// TransformerBlocks_10857677324520
// MI455X (gfx1250) — compile-verified, matrix operand order checked by reading, not run on hardware
//
#include <hip/hip_runtime.h>
#include <hip/hip_bf16.h>
#include <math.h>

#define B_ 2
#define S_ 2048
#define D_ 1024
#define H_ 16
#define HD_ 64
#define DFF_ 4096

typedef __attribute__((ext_vector_type(16))) __bf16 v16bf;
typedef __attribute__((ext_vector_type(8))) float v8f;

union FragU { uint4 u[2]; v16bf v; };

__device__ __forceinline__ unsigned short f2bf(float x) {
  unsigned int u = __float_as_uint(x);
  u += 0x7FFFu + ((u >> 16) & 1u);
  return (unsigned short)(u >> 16);
}

// Load one 16x32 bf16 fragment (A layout; Bt uses the same layout with N as rows).
// Per ISA: lanes 0-15 row=lane, K chunks [k0+half*8 .. +7] and [k0+16+half*8 .. +7].
__device__ __forceinline__ v16bf load_frag(const unsigned short* p, int ld, int row0, int k0) {
  int lane = threadIdx.x & 31;
  int hf = lane >> 4;
  int r = lane & 15;
  const unsigned short* base = p + (size_t)(row0 + r) * ld + k0 + hf * 8;
  FragU f;
  f.u[0] = *(const uint4*)(base);
  f.u[1] = *(const uint4*)(base + 16);
  return f.v;
}

// Same, but from a precomputed per-lane base pointer (row/half already applied).
__device__ __forceinline__ v16bf load_frag_ptr(const unsigned short* lanebase, int k0) {
  FragU f;
  f.u[0] = *(const uint4*)(lanebase + k0);
  f.u[1] = *(const uint4*)(lanebase + k0 + 16);
  return f.v;
}

__device__ __forceinline__ v8f wmma_bf16(v16bf a, v16bf b, v8f c) {
  return __builtin_amdgcn_wmma_f32_16x16x32_bf16(false, a, false, b, (short)0, c, false, false);
}

// ---------------------------------------------------------------------------
// Weight convert + transpose: in f32 [K,N] -> out bf16 [N,K]
__global__ void cvt_transpose(const float* __restrict__ in, unsigned short* __restrict__ out,
                              int K, int N) {
  unsigned int i = blockIdx.x * blockDim.x + threadIdx.x;
  if (i >= (unsigned int)(K * N)) return;
  int n = i / K;
  int k = i % K;
  out[i] = f2bf(in[(size_t)k * N + n]);
}

// ---------------------------------------------------------------------------
// LayerNorm over D=1024, one block (256 threads) per row, bf16 output.
__global__ __launch_bounds__(256) void layernorm_bf16(
    const float* __restrict__ x, const float* __restrict__ scale,
    const float* __restrict__ shift, unsigned short* __restrict__ out) {
  int row = blockIdx.x;
  int t = threadIdx.x;
  const float* xr = x + (size_t)row * D_;
  float4 v = *(const float4*)(xr + t * 4);
  float s = v.x + v.y + v.z + v.w;
  float sq = v.x * v.x + v.y * v.y + v.z * v.z + v.w * v.w;
  for (int off = 1; off < 32; off <<= 1) {
    s += __shfl_xor(s, off, 32);
    sq += __shfl_xor(sq, off, 32);
  }
  __shared__ float ss[8], ssq[8];
  __shared__ float mean_s, rstd_s;
  int wave = t >> 5;
  if ((t & 31) == 0) { ss[wave] = s; ssq[wave] = sq; }
  __syncthreads();
  if (t == 0) {
    float S = 0.f, Q = 0.f;
    for (int i = 0; i < 8; ++i) { S += ss[i]; Q += ssq[i]; }
    float mean = S * (1.0f / D_);
    float var = Q * (1.0f / D_) - mean * mean;
    mean_s = mean;
    rstd_s = rsqrtf(var + 1e-5f);
  }
  __syncthreads();
  float mean = mean_s, rstd = rstd_s;
  float y0 = (v.x - mean) * rstd * scale[t * 4 + 0] + shift[t * 4 + 0];
  float y1 = (v.y - mean) * rstd * scale[t * 4 + 1] + shift[t * 4 + 1];
  float y2 = (v.z - mean) * rstd * scale[t * 4 + 2] + shift[t * 4 + 2];
  float y3 = (v.w - mean) * rstd * scale[t * 4 + 3] + shift[t * 4 + 3];
  uint2 o;
  o.x = (unsigned)f2bf(y0) | ((unsigned)f2bf(y1) << 16);
  o.y = (unsigned)f2bf(y2) | ((unsigned)f2bf(y3) << 16);
  *(uint2*)(out + (size_t)row * D_ + t * 4) = o;
}

// ---------------------------------------------------------------------------
// GEMM: C[M,N] = A[M,K](bf16) x Bt[N,K](bf16)^T, fused epilogue.
// Block = 8 waves arranged 4(M) x 2(N); each wave computes a 32x32 tile with
// 4 independent WMMA accumulator chains (hides WMMA RAW latency) and 2x
// fragment reuse. Block tile = 128 x 64.
// flags: 1=bias, 2=gelu, 4=add residual(f32), 8=store f32, 16=store bf16
__global__ __launch_bounds__(256) void gemm_bf16(
    const unsigned short* __restrict__ A, const unsigned short* __restrict__ Bt,
    const float* __restrict__ bias, const float* __restrict__ res,
    float* __restrict__ Cf, unsigned short* __restrict__ Cb,
    int M, int N, int K, int flags) {
  int wave = threadIdx.x >> 5;
  int lane = threadIdx.x & 31;
  int hf = lane >> 4, r = lane & 15;
  int m0 = blockIdx.y * 128 + (wave & 3) * 32;
  int n0 = blockIdx.x * 64 + (wave >> 2) * 32;

  // Per-lane base pointers (row + K-chunk half offset baked in).
  const unsigned short* pa0 = A + (size_t)(m0 + r) * K + hf * 8;
  const unsigned short* pa1 = pa0 + (size_t)16 * K;
  const unsigned short* pb0 = Bt + (size_t)(n0 + r) * K + hf * 8;
  const unsigned short* pb1 = pb0 + (size_t)16 * K;

  v8f acc00 = {}, acc01 = {}, acc10 = {}, acc11 = {};
  for (int k0 = 0; k0 < K; k0 += 32) {
    if (k0 + 128 < K) {  // stream-ahead prefetch (global_prefetch_b8)
      __builtin_prefetch(pa0 + k0 + 128, 0, 1);
      __builtin_prefetch(pb0 + k0 + 128, 0, 1);
    }
    v16bf a0 = load_frag_ptr(pa0, k0);
    v16bf a1 = load_frag_ptr(pa1, k0);
    v16bf b0 = load_frag_ptr(pb0, k0);
    v16bf b1 = load_frag_ptr(pb1, k0);
    acc00 = wmma_bf16(a0, b0, acc00);
    acc01 = wmma_bf16(a0, b1, acc01);
    acc10 = wmma_bf16(a1, b0, acc10);
    acc11 = wmma_bf16(a1, b1, acc11);
  }

  float bv0 = (flags & 1) ? bias[n0 + r] : 0.0f;
  float bv1 = (flags & 1) ? bias[n0 + 16 + r] : 0.0f;
  v8f accs[2][2] = {{acc00, acc01}, {acc10, acc11}};
#pragma unroll
  for (int im = 0; im < 2; ++im) {
#pragma unroll
    for (int in2 = 0; in2 < 2; ++in2) {
      v8f acc = accs[im][in2];
      int col = n0 + in2 * 16 + r;
      float bv = in2 ? bv1 : bv0;
#pragma unroll
      for (int j = 0; j < 8; ++j) {
        int row = m0 + im * 16 + j + hf * 8;
        float v = acc[j] + bv;
        if (flags & 2) {
          float t = v;
          v = 0.5f * t * (1.0f + tanhf(0.7978845608028654f * (t + 0.044715f * t * t * t)));
        }
        size_t idx = (size_t)row * N + col;
        if (flags & 4) v += res[idx];
        if (flags & 8) Cf[idx] = v;
        if (flags & 16) Cb[idx] = f2bf(v);
      }
    }
  }
}

// ---------------------------------------------------------------------------
// v bf16 [B*S, D] -> vt bf16 [B*H, HD, S]  (V^T per head, so ctx GEMM Bt is contiguous)
__global__ void transpose_v(const unsigned short* __restrict__ v, unsigned short* __restrict__ vt) {
  unsigned int i = blockIdx.x * blockDim.x + threadIdx.x;
  if (i >= (unsigned int)(B_ * S_ * D_)) return;
  int d = i % D_;
  unsigned int bs = i / D_;
  int s = bs % S_;
  int b = bs / S_;
  int h = d / HD_, hd = d % HD_;
  vt[((size_t)(b * H_ + h) * HD_ + hd) * S_ + s] = v[i];
}

// ---------------------------------------------------------------------------
// Flash-style causal attention: one wave per (b, h, 16 query rows).
__global__ __launch_bounds__(256) void flash_attn(
    const unsigned short* __restrict__ q, const unsigned short* __restrict__ k,
    const unsigned short* __restrict__ vt, unsigned short* __restrict__ ctx) {
  __shared__ __align__(16) unsigned short psh[8 * 16 * 32];
  int wave = threadIdx.x >> 5;
  int lane = threadIdx.x & 31;
  int hf = lane >> 4, r = lane & 15;
  int task = blockIdx.x * 8 + wave;
  int qt = task % (S_ / 16);
  int bh = task / (S_ / 16);
  int h = bh % H_;
  int b = bh / H_;
  int qbase = qt * 16;
  const unsigned short* qp = q + (size_t)b * S_ * D_ + h * HD_;
  const unsigned short* kp = k + (size_t)b * S_ * D_ + h * HD_;
  const unsigned short* vp = vt + (size_t)(b * H_ + h) * HD_ * S_;
  unsigned short* pme = psh + wave * 512;

  v16bf aq0 = load_frag(qp, D_, qbase, 0);
  v16bf aq1 = load_frag(qp, D_, qbase, 32);
  v8f acc0 = {}, acc1 = {}, acc2 = {}, acc3 = {};
  float mrun[8], lrun[8];
#pragma unroll
  for (int j = 0; j < 8; ++j) { mrun[j] = -1e30f; lrun[j] = 0.0f; }

  int kend = qbase + 15;
  for (int kb = 0; kb <= kend; kb += 32) {
    v8f s0 = {}, s1 = {};
    s0 = wmma_bf16(aq0, load_frag(kp, D_, kb, 0), s0);
    s0 = wmma_bf16(aq1, load_frag(kp, D_, kb, 32), s0);
    s1 = wmma_bf16(aq0, load_frag(kp, D_, kb + 16, 0), s1);
    s1 = wmma_bf16(aq1, load_frag(kp, D_, kb + 16, 32), s1);
#pragma unroll
    for (int j = 0; j < 8; ++j) {
      int qr = qbase + j + hf * 8;
      float x0 = s0[j] * 0.125f;   // 1/sqrt(64)
      float x1 = s1[j] * 0.125f;
      if (kb + r > qr) x0 = -1e30f;
      if (kb + 16 + r > qr) x1 = -1e30f;
      float mloc = fmaxf(x0, x1);
      for (int off = 1; off < 16; off <<= 1) mloc = fmaxf(mloc, __shfl_xor(mloc, off, 32));
      float mnew = fmaxf(mrun[j], mloc);
      float p0 = __expf(x0 - mnew);
      float p1 = __expf(x1 - mnew);
      float sl = p0 + p1;
      for (int off = 1; off < 16; off <<= 1) sl += __shfl_xor(sl, off, 32);
      float resc = __expf(mrun[j] - mnew);
      lrun[j] = lrun[j] * resc + sl;
      mrun[j] = mnew;
      acc0[j] *= resc; acc1[j] *= resc; acc2[j] *= resc; acc3[j] *= resc;
      int prow = j + hf * 8;
      pme[prow * 32 + r] = f2bf(p0);
      pme[prow * 32 + 16 + r] = f2bf(p1);
    }
    v16bf ap = load_frag(pme, 32, 0, 0);     // LDS read, A-fragment of P (16x32)
    acc0 = wmma_bf16(ap, load_frag(vp, S_, 0, kb), acc0);
    acc1 = wmma_bf16(ap, load_frag(vp, S_, 16, kb), acc1);
    acc2 = wmma_bf16(ap, load_frag(vp, S_, 32, kb), acc2);
    acc3 = wmma_bf16(ap, load_frag(vp, S_, 48, kb), acc3);
  }
  unsigned short* cp = ctx + (size_t)b * S_ * D_ + h * HD_;
#pragma unroll
  for (int j = 0; j < 8; ++j) {
    float inv = 1.0f / lrun[j];
    size_t base = (size_t)(qbase + j + hf * 8) * D_;
    cp[base + 0 + r] = f2bf(acc0[j] * inv);
    cp[base + 16 + r] = f2bf(acc1[j] * inv);
    cp[base + 32 + r] = f2bf(acc2[j] * inv);
    cp[base + 48 + r] = f2bf(acc3[j] * inv);
  }
}

// ---------------------------------------------------------------------------
extern "C" void kernel_launch(void* const* d_in, const int* in_sizes, int n_in,
                              void* d_out, int out_size, void* d_ws, size_t ws_size,
                              hipStream_t stream) {
  (void)in_sizes; (void)n_in; (void)out_size; (void)ws_size;
  const float* x    = (const float*)d_in[0];
  const float* ln1s = (const float*)d_in[1];
  const float* ln1b = (const float*)d_in[2];
  const float* wq   = (const float*)d_in[3];
  const float* wk   = (const float*)d_in[4];
  const float* wv   = (const float*)d_in[5];
  const float* wo   = (const float*)d_in[6];
  const float* bo   = (const float*)d_in[7];
  const float* ln2s = (const float*)d_in[8];
  const float* ln2b = (const float*)d_in[9];
  const float* w1   = (const float*)d_in[10];
  const float* b1   = (const float*)d_in[11];
  const float* w2   = (const float*)d_in[12];
  const float* b2   = (const float*)d_in[13];
  float* out = (float*)d_out;

  size_t off = 0;
  auto alloc = [&](size_t bytes) {
    void* p = (char*)d_ws + off;
    off += (bytes + 255) & ~(size_t)255;
    return p;
  };
  unsigned short* wqT  = (unsigned short*)alloc((size_t)D_ * D_ * 2);
  unsigned short* wkT  = (unsigned short*)alloc((size_t)D_ * D_ * 2);
  unsigned short* wvT  = (unsigned short*)alloc((size_t)D_ * D_ * 2);
  unsigned short* woT  = (unsigned short*)alloc((size_t)D_ * D_ * 2);
  unsigned short* w1T  = (unsigned short*)alloc((size_t)D_ * DFF_ * 2);
  unsigned short* w2T  = (unsigned short*)alloc((size_t)DFF_ * D_ * 2);
  unsigned short* ln1x = (unsigned short*)alloc((size_t)B_ * S_ * D_ * 2);
  unsigned short* qb   = (unsigned short*)alloc((size_t)B_ * S_ * D_ * 2);
  unsigned short* kb   = (unsigned short*)alloc((size_t)B_ * S_ * D_ * 2);
  unsigned short* vb   = (unsigned short*)alloc((size_t)B_ * S_ * D_ * 2);
  unsigned short* vtb  = (unsigned short*)alloc((size_t)B_ * S_ * D_ * 2);
  unsigned short* ctxb = (unsigned short*)alloc((size_t)B_ * S_ * D_ * 2);
  float*          hbuf = (float*)alloc((size_t)B_ * S_ * D_ * 4);
  unsigned short* ln2x = (unsigned short*)alloc((size_t)B_ * S_ * D_ * 2);
  unsigned short* midb = (unsigned short*)alloc((size_t)B_ * S_ * DFF_ * 2);

  const int M = B_ * S_;

  // Weight prep (bf16 transposed)
  {
    unsigned int nDD = (unsigned int)D_ * D_;
    unsigned int g = (nDD + 255) / 256;
    cvt_transpose<<<g, 256, 0, stream>>>(wq, wqT, D_, D_);
    cvt_transpose<<<g, 256, 0, stream>>>(wk, wkT, D_, D_);
    cvt_transpose<<<g, 256, 0, stream>>>(wv, wvT, D_, D_);
    cvt_transpose<<<g, 256, 0, stream>>>(wo, woT, D_, D_);
    unsigned int nDF = (unsigned int)D_ * DFF_;
    cvt_transpose<<<(nDF + 255) / 256, 256, 0, stream>>>(w1, w1T, D_, DFF_);
    cvt_transpose<<<(nDF + 255) / 256, 256, 0, stream>>>(w2, w2T, DFF_, D_);
  }

  // LN1
  layernorm_bf16<<<M, 256, 0, stream>>>(x, ln1s, ln1b, ln1x);

  // Q, K, V projections (bf16 outputs)
  dim3 gDD(D_ / 64, M / 128);
  gemm_bf16<<<gDD, 256, 0, stream>>>(ln1x, wqT, nullptr, nullptr, nullptr, qb, M, D_, D_, 16);
  gemm_bf16<<<gDD, 256, 0, stream>>>(ln1x, wkT, nullptr, nullptr, nullptr, kb, M, D_, D_, 16);
  gemm_bf16<<<gDD, 256, 0, stream>>>(ln1x, wvT, nullptr, nullptr, nullptr, vb, M, D_, D_, 16);

  // V^T per head
  transpose_v<<<((unsigned)(B_ * S_ * D_) + 255) / 256, 256, 0, stream>>>(vb, vtb);

  // Causal flash attention -> ctx (bf16)
  flash_attn<<<(B_ * H_ * (S_ / 16)) / 8, 256, 0, stream>>>(qb, kb, vtb, ctxb);

  // h = x + ctx @ wo + bo  (f32)
  gemm_bf16<<<gDD, 256, 0, stream>>>(ctxb, woT, bo, x, hbuf, nullptr, M, D_, D_, 1 | 4 | 8);

  // LN2
  layernorm_bf16<<<M, 256, 0, stream>>>(hbuf, ln2s, ln2b, ln2x);

  // FFN1: gelu(ln2 @ w1 + b1) -> bf16
  dim3 gDF(DFF_ / 64, M / 128);
  gemm_bf16<<<gDF, 256, 0, stream>>>(ln2x, w1T, b1, nullptr, nullptr, midb, M, DFF_, D_, 1 | 2 | 16);

  // out = h + mid @ w2 + b2  (f32)
  gemm_bf16<<<gDD, 256, 0, stream>>>(midb, w2T, b2, hbuf, out, nullptr, M, D_, DFF_, 1 | 4 | 8);
}
